// Block_8272107012565
// MI455X (gfx1250) — matrix operand order repaired and checked by reading, not compiled, not run on hardware
//
#include <hip/hip_runtime.h>
#include <hip/hip_bf16.h>
#include <math.h>

typedef unsigned short u16;
typedef __attribute__((ext_vector_type(16))) __bf16 v16bf;
typedef __attribute__((ext_vector_type(8)))  float  v8f;

#define BATCH   64
#define SEQ     197
#define DMODEL  768
#define NHEADS  12
#define DHEAD   64
#define FFDIM   3072
#define MTOK    (BATCH*SEQ)   // 12608 = 197*64
#define SPAD    208           // 13*16  (q,k rows; score rows/cols)
#define SPADV   224           // 7*32   (v rows; probs cols)
#define NEG_BIG (-3.0e38f)

#define USE_ASYNC_LDS 1

// ---------- helpers ----------
static __device__ __forceinline__ u16 f2bf(float f) {
  unsigned int u = __builtin_bit_cast(unsigned int, f);
  u += 0x7FFFu + ((u >> 16) & 1u);          // round-to-nearest-even
  return (u16)(u >> 16);
}

static __device__ __forceinline__ v16bf mkfrag(uint4 lo, uint4 hi) {
  union { uint4 u[2]; v16bf v; } c;
  c.u[0] = lo; c.u[1] = hi;
  return c.v;
}

// LDS byte offset of a __shared__ object (addrspace(3) pointers are LDS-relative)
static __device__ __forceinline__ unsigned lds_off(const void* p) {
  return (unsigned)(unsigned long long)(__attribute__((address_space(3))) const char*)p;
}

#if USE_ASYNC_LDS
// async global -> LDS 16-byte copy (GV mode), tracked by ASYNCcnt
static __device__ __forceinline__ void async_copy16(const u16* g, unsigned lds_byte) {
  asm volatile("global_load_async_to_lds_b128 %0, %1, off"
               :: "v"(lds_byte), "v"((unsigned long long)(size_t)g)
               : "memory");
}
static __device__ __forceinline__ void wait_async0() {
#if __has_builtin(__builtin_amdgcn_s_wait_asynccnt)
  __builtin_amdgcn_s_wait_asynccnt(0);
#else
  asm volatile("s_wait_asynccnt 0" ::: "memory");
#endif
}
#endif

// ---------- elementwise utility kernels ----------
__global__ void k_cvt_bf16(const float* __restrict__ src, u16* __restrict__ dst, int n) {
  int i = blockIdx.x * blockDim.x + threadIdx.x;
  if (i < n) dst[i] = f2bf(src[i]);
}

__global__ void k_zero_u16(u16* __restrict__ p, int n) {
  int i = blockIdx.x * blockDim.x + threadIdx.x;
  if (i < n) p[i] = 0;
}

// ---------- layernorm: fp32 [row][768] -> bf16 ----------
__global__ __launch_bounds__(256)
void k_layernorm(const float* __restrict__ x, const float* __restrict__ w,
                 const float* __restrict__ b, u16* __restrict__ out) {
  __shared__ float red[256];
  const int tid = threadIdx.x;
  const int row = blockIdx.x;
  const float* xr = x + (size_t)row * DMODEL;
  float v0 = xr[tid], v1 = xr[tid + 256], v2 = xr[tid + 512];
  red[tid] = v0 + v1 + v2;
  __syncthreads();
  for (int st = 128; st > 0; st >>= 1) { if (tid < st) red[tid] += red[tid + st]; __syncthreads(); }
  const float mean = red[0] * (1.0f / DMODEL);
  __syncthreads();
  float d0 = v0 - mean, d1 = v1 - mean, d2 = v2 - mean;
  red[tid] = d0 * d0 + d1 * d1 + d2 * d2;
  __syncthreads();
  for (int st = 128; st > 0; st >>= 1) { if (tid < st) red[tid] += red[tid + st]; __syncthreads(); }
  const float rstd = rsqrtf(red[0] * (1.0f / DMODEL) + 1e-6f);
  u16* orow = out + (size_t)row * DMODEL;
  orow[tid]       = f2bf(d0 * rstd * w[tid]       + b[tid]);
  orow[tid + 256] = f2bf(d1 * rstd * w[tid + 256] + b[tid + 256]);
  orow[tid + 512] = f2bf(d2 * rstd * w[tid + 512] + b[tid + 512]);
}

#define TK 32

// ================= big WMMA GEMM: 8 waves, 64x128 tile, wave = 32x32 (4 WMMA) =================
// MODE 1: bf16 scatter to [b,h,spad,64] (qkv)     (bias)
// MODE 2: gelu -> bf16 [m][ldo]                    (bias)
// MODE 3: fp32 out[m][ldo] = acc + bias + res[m][ldo]
#define BTM 64
#define BTN 128

template<int MODE>
__global__ __launch_bounds__(256)
void k_gemm_big(const u16* __restrict__ A, int lda,
                const u16* __restrict__ Bm, int ldb,
                const float* __restrict__ bias, int K,
                u16* __restrict__ outB, float* __restrict__ outF,
                const float* __restrict__ resF, int ldo, int spad) {
  __shared__ u16 lsA[BTM * TK];   // A tile row-major [64][32]
  __shared__ u16 lsB[BTN * TK];   // B tile transposed [n][k] = [128][32]
  const int tid  = threadIdx.x;
  const int w    = tid >> 5;
  const int lane = tid & 31;
  const int half = lane >> 4;
  const int lr   = lane & 15;
  const int wm   = w & 1;          // 2 M-groups of 32
  const int wn   = w >> 1;         // 4 N-groups of 32
  const int mBase = blockIdx.x * BTM;
  const int nBase = blockIdx.y * BTN;

  const int arow = tid >> 2, acol = (tid & 3) * 8;   // A staging: 16B/thread
  const int bkk  = tid >> 3, bnc  = (tid & 7) * 16;  // B staging: 32B/thread
#if USE_ASYNC_LDS
  const unsigned ldsA_byte = lds_off(lsA) + (unsigned)(arow * TK + acol) * 2u;
#endif

  v8f acc00 = {}, acc01 = {}, acc10 = {}, acc11 = {};
  for (int k0 = 0; k0 < K; k0 += TK) {
    // stage A (64x32): one async 16B chunk per thread
#if USE_ASYNC_LDS
    async_copy16(A + (size_t)(mBase + arow) * lda + k0 + acol, ldsA_byte);
#else
    *(uint4*)(lsA + arow * TK + acol) =
        *(const uint4*)(A + (size_t)(mBase + arow) * lda + k0 + acol);
#endif
    // stage B (32x128) transposed: 16 elems per thread
    {
      const u16* src = Bm + (size_t)(k0 + bkk) * ldb + nBase + bnc;
      u16 tmp[16];
      *(uint4*)(tmp)     = *(const uint4*)(src);
      *(uint4*)(tmp + 8) = *(const uint4*)(src + 8);
      #pragma unroll
      for (int j = 0; j < 16; ++j) lsB[(bnc + j) * TK + bkk] = tmp[j];
    }
    if (k0 + TK < K)
      __builtin_prefetch(Bm + (size_t)(k0 + TK + bkk) * ldb + nBase + bnc, 0, 1);
#if USE_ASYNC_LDS
    wait_async0();
#endif
    __syncthreads();
    // fragments: A 16x32 layout (k halves of 8), B 32x16 layout (k halves of 8)
    const int am0 = (wm * 32 + lr) * TK;
    const int am1 = (wm * 32 + 16 + lr) * TK;
    const v16bf a0 = mkfrag(*(const uint4*)(lsA + am0 + 8 * half),
                            *(const uint4*)(lsA + am0 + 16 + 8 * half));
    const v16bf a1 = mkfrag(*(const uint4*)(lsA + am1 + 8 * half),
                            *(const uint4*)(lsA + am1 + 16 + 8 * half));
    const int bn0 = (wn * 32 + lr) * TK + 8 * half;
    const int bn1 = (wn * 32 + 16 + lr) * TK + 8 * half;
    const v16bf b0 = mkfrag(*(const uint4*)(lsB + bn0), *(const uint4*)(lsB + bn0 + 16));
    const v16bf b1 = mkfrag(*(const uint4*)(lsB + bn1), *(const uint4*)(lsB + bn1 + 16));
    acc00 = __builtin_amdgcn_wmma_f32_16x16x32_bf16(false, a0, false, b0, (short)0, acc00, false, false);
    acc01 = __builtin_amdgcn_wmma_f32_16x16x32_bf16(false, a0, false, b1, (short)0, acc01, false, false);
    acc10 = __builtin_amdgcn_wmma_f32_16x16x32_bf16(false, a1, false, b0, (short)0, acc10, false, false);
    acc11 = __builtin_amdgcn_wmma_f32_16x16x32_bf16(false, a1, false, b1, (short)0, acc11, false, false);
    __syncthreads();
  }

  #pragma unroll
  for (int mi = 0; mi < 2; ++mi) {
    #pragma unroll
    for (int ni = 0; ni < 2; ++ni) {
      const v8f acc = (mi == 0) ? (ni == 0 ? acc00 : acc01) : (ni == 0 ? acc10 : acc11);
      const int nGlob = nBase + wn * 32 + ni * 16 + lr;
      const float bval = bias ? bias[nGlob] : 0.0f;
      #pragma unroll
      for (int r = 0; r < 8; ++r) {
        const int m = mBase + wm * 32 + mi * 16 + half * 8 + r;
        const float v = acc[r] + bval;
        if (MODE == 1) {
          const int t = m, bb = t / SEQ, s = t % SEQ;
          const int h = nGlob >> 6, dh = nGlob & 63;
          outB[(((size_t)bb * NHEADS + h) * spad + s) * DHEAD + dh] = f2bf(v);
        } else if (MODE == 2) {
          const float g = 0.5f * v * (1.0f + erff(v * 0.70710678118f));
          outB[(size_t)m * ldo + nGlob] = f2bf(g);
        } else if (MODE == 3) {
          outF[(size_t)m * ldo + nGlob] = v + resF[(size_t)m * ldo + nGlob];
        }
      }
    }
  }
}

// ================= small WMMA GEMM: 4 waves, 16x64 tile (batched attn@v) =================
// MODE 4: batched attn@v: bf16 to [tok][768] scatter (no bias), z = b*12+h
template<int MODE>
__global__ __launch_bounds__(128)
void k_gemm(const u16* __restrict__ A, int lda, unsigned long long strideAz,
            const u16* __restrict__ Bm, int ldb, unsigned long long strideBz,
            const float* __restrict__ bias, int K,
            u16* __restrict__ outB, float* __restrict__ outF,
            const float* __restrict__ resF, int ldo, int spad) {
  __shared__ u16 lsA[16 * TK];
  __shared__ u16 lsB[64 * TK];
  const int tid  = threadIdx.x;
  const int w    = tid >> 5;
  const int lane = tid & 31;
  const int half = lane >> 4;
  const int lr   = lane & 15;
  const int mBase = blockIdx.x * 16;
  const int nBase = blockIdx.y * 64;
  const int z     = blockIdx.z;
  const u16* Ab = A  + (size_t)z * strideAz;
  const u16* Bb = Bm + (size_t)z * strideBz;

  v8f acc = {};
  for (int k0 = 0; k0 < K; k0 += TK) {
    {
      const int row = tid >> 3, col = (tid & 7) * 4;
      *(uint2*)(lsA + row * TK + col) =
          *(const uint2*)(Ab + (size_t)(mBase + row) * lda + k0 + col);
    }
    {
      const int kk = tid >> 2, nc = (tid & 3) * 16;
      const u16* src = Bb + (size_t)(k0 + kk) * ldb + nBase + nc;
      u16 tmp[16];
      *(uint4*)(tmp)     = *(const uint4*)(src);
      *(uint4*)(tmp + 8) = *(const uint4*)(src + 8);
      #pragma unroll
      for (int j = 0; j < 16; ++j) lsB[(nc + j) * TK + kk] = tmp[j];
    }
    __syncthreads();
    const uint4 alo = *(const uint4*)(lsA + lr * TK + 8 * half);
    const uint4 ahi = *(const uint4*)(lsA + lr * TK + 16 + 8 * half);
    const int bcol = w * 16 + lr;
    const uint4 blo = *(const uint4*)(lsB + bcol * TK + 8 * half);
    const uint4 bhi = *(const uint4*)(lsB + bcol * TK + 16 + 8 * half);
    acc = __builtin_amdgcn_wmma_f32_16x16x32_bf16(
        false, mkfrag(alo, ahi), false, mkfrag(blo, bhi), (short)0, acc, false, false);
    __syncthreads();
  }

  const int nGlob = nBase + w * 16 + lr;
  const float bval = bias ? bias[nGlob] : 0.0f;
  #pragma unroll
  for (int r = 0; r < 8; ++r) {
    const int m = mBase + half * 8 + r;
    const float v = acc[r] + bval;
    if (MODE == 3) {
      outF[(size_t)m * ldo + nGlob] = v + resF[(size_t)m * ldo + nGlob];
    } else if (MODE == 4) {
      const int s = m;
      if (s < SEQ) {
        const int bb = z / NHEADS, h = z % NHEADS;
        outB[((size_t)bb * SEQ + s) * DMODEL + h * DHEAD + nGlob] = f2bf(v);
      }
    }
  }
  (void)spad;
}

// ---------- attention scores: q@k^T (both fragments straight from global) ----------
__global__ __launch_bounds__(128)
void k_scores(const u16* __restrict__ q, const u16* __restrict__ k,
              const int* __restrict__ mask, float* __restrict__ scores) {
  const int bh   = blockIdx.x;
  const int w    = threadIdx.x >> 5;
  const int lane = threadIdx.x & 31;
  const int half = lane >> 4;
  const int lr   = lane & 15;
  const int t = blockIdx.y * 4 + w;      // tile id, wave-uniform
  if (t >= 169) return;                  // 13x13 tiles of 16x16
  const int mi = t / 13, ni = t % 13;
  const u16* qb = q + (size_t)bh * SPAD * DHEAD;
  const u16* kb = k + (size_t)bh * SPAD * DHEAD;
  v8f acc = {};
  #pragma unroll
  for (int kc = 0; kc < DHEAD; kc += 32) {
    const int arow = mi * 16 + lr;
    const uint4 alo = *(const uint4*)(qb + (size_t)arow * DHEAD + kc + 8 * half);
    const uint4 ahi = *(const uint4*)(qb + (size_t)arow * DHEAD + kc + 16 + 8 * half);
    const int bcol = ni * 16 + lr;       // column of k^T == row of k: contiguous
    const uint4 blo = *(const uint4*)(kb + (size_t)bcol * DHEAD + kc + 8 * half);
    const uint4 bhi = *(const uint4*)(kb + (size_t)bcol * DHEAD + kc + 16 + 8 * half);
    acc = __builtin_amdgcn_wmma_f32_16x16x32_bf16(
        false, mkfrag(alo, ahi), false, mkfrag(blo, bhi), (short)0, acc, false, false);
  }
  const int b    = bh / NHEADS;
  const int ncol = ni * 16 + lr;
  const float mb = (ncol < SEQ) ? (-10000.0f * (1.0f - (float)mask[b * SEQ + ncol])) : NEG_BIG;
  float* sb = scores + (size_t)bh * SPAD * SPAD;
  #pragma unroll
  for (int r = 0; r < 8; ++r) {
    const int m = mi * 16 + half * 8 + r;
    sb[(size_t)m * SPAD + ncol] = acc[r] * 0.125f + mb;   // 1/sqrt(64)
  }
}

// ---------- softmax over 208 cols -> bf16 probs [208][224] ----------
__global__ __launch_bounds__(256)
void k_softmax(const float* __restrict__ scores, u16* __restrict__ probs) {
  __shared__ float red[256];
  const int bh = blockIdx.x, row = blockIdx.y, tid = threadIdx.x;
  u16* prow = probs + ((size_t)bh * SPAD + row) * SPADV;
  if (row >= SEQ) { if (tid < SPADV) prow[tid] = 0; return; }
  const float* srow = scores + ((size_t)bh * SPAD + row) * SPAD;
  const float s = (tid < SPAD) ? srow[tid] : NEG_BIG;
  red[tid] = s;
  __syncthreads();
  for (int st = 128; st > 0; st >>= 1) { if (tid < st) red[tid] = fmaxf(red[tid], red[tid + st]); __syncthreads(); }
  const float mx = red[0];
  __syncthreads();
  const float e = (tid < SPAD) ? __expf(s - mx) : 0.0f;
  red[tid] = e;
  __syncthreads();
  for (int st = 128; st > 0; st >>= 1) { if (tid < st) red[tid] += red[tid + st]; __syncthreads(); }
  const float inv = 1.0f / red[0];
  if (tid < SPAD)        prow[tid] = f2bf(e * inv);
  else if (tid < SPADV)  prow[tid] = 0;
}

// ---------- host ----------
extern "C" void kernel_launch(void* const* d_in, const int* in_sizes, int n_in,
                              void* d_out, int out_size, void* d_ws, size_t ws_size,
                              hipStream_t stream) {
  (void)in_sizes; (void)n_in; (void)out_size; (void)ws_size;
  const float* x    = (const float*)d_in[0];
  const float* Wq   = (const float*)d_in[1];  const float* bq = (const float*)d_in[2];
  const float* Wk   = (const float*)d_in[3];  const float* bk = (const float*)d_in[4];
  const float* Wv   = (const float*)d_in[5];  const float* bv = (const float*)d_in[6];
  const float* Wo   = (const float*)d_in[7];  const float* bo = (const float*)d_in[8];
  const float* ln1w = (const float*)d_in[9];  const float* ln1b = (const float*)d_in[10];
  const float* W1   = (const float*)d_in[11]; const float* b1 = (const float*)d_in[12];
  const float* W2   = (const float*)d_in[13]; const float* b2 = (const float*)d_in[14];
  const float* ln2w = (const float*)d_in[15]; const float* ln2b = (const float*)d_in[16];
  const int*   mask = (const int*)d_in[17];
  float* out = (float*)d_out;

  size_t off = 0;
  auto take = [&](size_t bytes) -> void* {
    off = (off + 255) & ~(size_t)255;
    void* p = (char*)d_ws + off;
    off += bytes;
    return p;
  };
  u16* wq  = (u16*)take((size_t)DMODEL * DMODEL * 2);
  u16* wk  = (u16*)take((size_t)DMODEL * DMODEL * 2);
  u16* wv  = (u16*)take((size_t)DMODEL * DMODEL * 2);
  u16* wo  = (u16*)take((size_t)DMODEL * DMODEL * 2);
  u16* w1  = (u16*)take((size_t)DMODEL * FFDIM * 2);
  u16* w2  = (u16*)take((size_t)FFDIM * DMODEL * 2);
  u16* hln = (u16*)take((size_t)MTOK * DMODEL * 2);
  const size_t nQK = (size_t)BATCH * NHEADS * SPAD * DHEAD;
  const size_t nV  = (size_t)BATCH * NHEADS * SPADV * DHEAD;
  u16* qb  = (u16*)take(nQK * 2);
  u16* kb  = (u16*)take(nQK * 2);
  u16* vb  = (u16*)take(nV * 2);
  u16* attn  = (u16*)take((size_t)MTOK * DMODEL * 2);
  u16* probs = (u16*)take((size_t)BATCH * NHEADS * SPAD * SPADV * 2);
  float* scores = (float*)take((size_t)BATCH * NHEADS * SPAD * SPAD * 4);
  u16* G = (u16*)scores;   // FFN1 activation reuses the (dead) scores region

  const dim3 blk128(128), blk256(256);
  auto cvt = [&](const float* s, u16* d, int n) {
    k_cvt_bf16<<<dim3((n + 255) / 256), blk256, 0, stream>>>(s, d, n);
  };
  cvt(Wq, wq, DMODEL * DMODEL);  cvt(Wk, wk, DMODEL * DMODEL);
  cvt(Wv, wv, DMODEL * DMODEL);  cvt(Wo, wo, DMODEL * DMODEL);
  cvt(W1, w1, DMODEL * FFDIM);   cvt(W2, w2, FFDIM * DMODEL);

  k_zero_u16<<<dim3((int)((nQK + 255) / 256)), blk256, 0, stream>>>(qb, (int)nQK);
  k_zero_u16<<<dim3((int)((nQK + 255) / 256)), blk256, 0, stream>>>(kb, (int)nQK);
  k_zero_u16<<<dim3((int)((nV  + 255) / 256)), blk256, 0, stream>>>(vb, (int)nV);

  // h = LN1(x)
  k_layernorm<<<dim3(MTOK), blk256, 0, stream>>>(x, ln1w, ln1b, hln);

  // q,k,v projections (64x128 tiles, scatter into padded [b,h,spad,64])
  const dim3 gProj(MTOK / BTM, DMODEL / BTN, 1);       // (197, 6)
  k_gemm_big<1><<<gProj, blk256, 0, stream>>>(hln, DMODEL, wq, DMODEL, bq, DMODEL,
                                              qb, nullptr, nullptr, 0, SPAD);
  k_gemm_big<1><<<gProj, blk256, 0, stream>>>(hln, DMODEL, wk, DMODEL, bk, DMODEL,
                                              kb, nullptr, nullptr, 0, SPAD);
  k_gemm_big<1><<<gProj, blk256, 0, stream>>>(hln, DMODEL, wv, DMODEL, bv, DMODEL,
                                              vb, nullptr, nullptr, 0, SPADV);

  // scores = q@k^T/8 + mask bias   (13x13 tiles per (b,h), 4 waves/block)
  k_scores<<<dim3(BATCH * NHEADS, 43), blk128, 0, stream>>>(qb, kb, mask, scores);

  // probs = softmax(scores), bf16, cols padded to 224
  k_softmax<<<dim3(BATCH * NHEADS, SPAD), blk256, 0, stream>>>(scores, probs);

  // attn = probs @ v  (batched over 768 heads), gather to [tok][768]
  k_gemm<4><<<dim3(SPAD / 16, 1, BATCH * NHEADS), blk128, 0, stream>>>(
      probs, SPADV, (unsigned long long)SPAD * SPADV,
      vb, DHEAD, (unsigned long long)SPADV * DHEAD,
      nullptr, SPADV, attn, nullptr, nullptr, 0, 0);

  // x1 = x + attn @ Wo + bo   -> d_out (fp32 residual stream)
  k_gemm_big<3><<<gProj, blk256, 0, stream>>>(attn, DMODEL, wo, DMODEL, bo, DMODEL,
                                              nullptr, out, x, DMODEL, 0);

  // h2 = LN2(x1)
  k_layernorm<<<dim3(MTOK), blk256, 0, stream>>>(out, ln2w, ln2b, hln);

  // G = gelu(h2 @ W1 + b1)
  k_gemm_big<2><<<dim3(MTOK / BTM, FFDIM / BTN, 1), blk256, 0, stream>>>(
      hln, DMODEL, w1, FFDIM, b1, DMODEL, G, nullptr, nullptr, FFDIM, 0);

  // out = x1 + G @ W2 + b2   (in-place residual on d_out)
  k_gemm_big<3><<<gProj, blk256, 0, stream>>>(G, FFDIM, w2, DMODEL, b2, FFDIM,
                                              nullptr, out, out, DMODEL, 0);
}
